// ChunkedSelfAttention_6777458393233
// MI455X (gfx1250) — matrix operand order repaired and checked by reading, not compiled, not run on hardware
//
#include <hip/hip_runtime.h>
#include <hip/hip_bf16.h>
#include <stdint.h>
#include <stddef.h>

// ---------------------------------------------------------------------------
// ChunkedSelfAttention for MI455X (gfx1250, wave32, WMMA, async global->LDS)
//   qkv = x @ w_attn^T ; rope(q,k) ; per-64-chunk softmax(QK^T/8)V ; y @ w_proj^T
// bf16 WMMA (v_wmma_f32_16x16x32_bf16) for all GEMMs, fp32 softmax/RoPE.
// GEMMs: double-buffered LDS staging fed by GLOBAL_LOAD_ASYNC_TO_LDS_B128.
// ---------------------------------------------------------------------------

typedef __attribute__((ext_vector_type(16))) __bf16 v16bf;
typedef __attribute__((ext_vector_type(8)))  __bf16 v8bf;
typedef __attribute__((ext_vector_type(8)))  float  v8f;

#define NHEAD 16
#define NEMBD 1024
#define CHUNK 64

static __device__ __forceinline__ __bf16 f2bf(float f) {
  uint32_t u = __builtin_bit_cast(uint32_t, f);
  uint32_t r = u + 0x7FFFu + ((u >> 16) & 1u);   // round-to-nearest-even
  uint16_t h = (uint16_t)(r >> 16);
  return __builtin_bit_cast(__bf16, h);
}
static __device__ __forceinline__ float bf2f(__bf16 b) {
  uint16_t h = __builtin_bit_cast(uint16_t, b);
  uint32_t u = ((uint32_t)h) << 16;
  return __builtin_bit_cast(float, u);
}
static __device__ __forceinline__ v8f v8f_zero() {
  v8f z = {0.f, 0.f, 0.f, 0.f, 0.f, 0.f, 0.f, 0.f};
  return z;
}
static __device__ __forceinline__ v8f wmma_bf16(v16bf a, v16bf b, v8f c) {
  return __builtin_amdgcn_wmma_f32_16x16x32_bf16(false, a, false, b,
                                                 (short)0, c, false, false);
}
static __device__ __forceinline__ void store_out(float* p, float v)  { *p = v; }
static __device__ __forceinline__ void store_out(__bf16* p, float v) { *p = f2bf(v); }

// ---- gfx1250 async global->LDS copy (ASYNCcnt-tracked, no VGPR round-trip) -
// VDST = LDS-relative byte offset, VADDR = 64-bit global VA (GV mode).
typedef __attribute__((address_space(3))) const void lds_cv;
static __device__ __forceinline__ uint32_t lds_offset(const void* p) {
  return (uint32_t)(uintptr_t)(lds_cv*)p;
}
static __device__ __forceinline__ void async_copy_b128(const void* gptr,
                                                       const void* lptr) {
  uint32_t l = lds_offset(lptr);
  uint64_t g = (uint64_t)(uintptr_t)gptr;
  asm volatile("global_load_async_to_lds_b128 %0, %1, off"
               :: "v"(l), "v"(g) : "memory");
}
static __device__ __forceinline__ void wait_async0() {
  asm volatile("s_wait_asynccnt 0" ::: "memory");
}

// --------------------------- fp32 -> bf16 convert ---------------------------
__global__ void cvt_f32_bf16(const float* __restrict__ in,
                             __bf16* __restrict__ out, int n) {
  int i = blockIdx.x * blockDim.x + threadIdx.x;
  int stride = gridDim.x * blockDim.x;
  for (; i < n; i += stride) out[i] = f2bf(in[i]);
}

// --------------------------- WMMA GEMM: C = A * W^T -------------------------
// A: M x K row-major bf16.  W: N x K row-major bf16 (so B[k][n] = W[n][k]).
// Block: 256 threads = 8 waves, tile 128(M) x 128(N); wave tile 32x64.
// K loop step 32; A/B K-slabs (128x32 bf16 = 8KB each) double-buffered in LDS,
// filled with async b128 copies (4 per thread per slab pair).
template <typename OutT>
__global__ void gemm_bf16_wmma(const __bf16* __restrict__ A,
                               const __bf16* __restrict__ W,
                               OutT* __restrict__ C,
                               int M, int N, int K) {
  __shared__ __attribute__((aligned(32))) __bf16 As[2][128][32];
  __shared__ __attribute__((aligned(32))) __bf16 Bs[2][128][32];

  const int tid   = threadIdx.x;
  const int lane  = tid & 31;
  const int w     = tid >> 5;        // 0..7
  const int waveM = w >> 1;          // 0..3 -> 32-row band
  const int waveN = w & 1;           // 0..1 -> 64-col band
  const int nBlk  = N >> 7;
  const int m0    = (blockIdx.x / nBlk) * 128;
  const int n0    = (blockIdx.x % nBlk) * 128;
  const int mr    = lane & 15;
  const int g     = lane >> 4;

  // staging assignment: 2 threads per row, 32B (2 x b128) per thread per slab
  const int ldRow  = tid >> 1;       // 0..127
  const int ldCol  = (tid & 1) * 16; // element offset 0 or 16
  const __bf16* agp = A + (size_t)(m0 + ldRow) * K + ldCol;
  const __bf16* wgp = W + (size_t)(n0 + ldRow) * K + ldCol;

  v8f acc[2][4];
#pragma unroll
  for (int i = 0; i < 2; i++)
#pragma unroll
    for (int j = 0; j < 4; j++) acc[i][j] = v8f_zero();

  // preload slab 0
#pragma unroll
  for (int c = 0; c < 2; c++) {
    async_copy_b128(agp + 8 * c, &As[0][ldRow][ldCol + 8 * c]);
    async_copy_b128(wgp + 8 * c, &Bs[0][ldRow][ldCol + 8 * c]);
  }
  wait_async0();
  __syncthreads();

  int buf = 0;
  for (int kb = 0; kb < K; kb += 32, buf ^= 1) {
    if (kb + 32 < K) {
#pragma unroll
      for (int c = 0; c < 2; c++) {
        async_copy_b128(agp + kb + 32 + 8 * c, &As[buf ^ 1][ldRow][ldCol + 8 * c]);
        async_copy_b128(wgp + kb + 32 + 8 * c, &Bs[buf ^ 1][ldRow][ldCol + 8 * c]);
      }
    }
    // A fragments for this wave's two 16-row tiles
    v16bf a0, a1;
    {
      const __bf16* r0 = &As[buf][32 * waveM + mr][0];
      const __bf16* r1 = &As[buf][32 * waveM + 16 + mr][0];
      v8bf lo = *(const v8bf*)(r0 + 16 * g);
      v8bf hi = *(const v8bf*)(r0 + 8 + 16 * g);
#pragma unroll
      for (int i = 0; i < 8; i++) { a0[i] = lo[i]; a0[8 + i] = hi[i]; }
      lo = *(const v8bf*)(r1 + 16 * g);
      hi = *(const v8bf*)(r1 + 8 + 16 * g);
#pragma unroll
      for (int i = 0; i < 8; i++) { a1[i] = lo[i]; a1[8 + i] = hi[i]; }
    }
#pragma unroll
    for (int j = 0; j < 4; j++) {
      v16bf b = *(const v16bf*)&Bs[buf][64 * waveN + 16 * j + mr][16 * g];
      acc[0][j] = wmma_bf16(a0, b, acc[0][j]);
      acc[1][j] = wmma_bf16(a1, b, acc[1][j]);
    }
    wait_async0();      // next slab landed (this wave's copies)
    __syncthreads();    // all waves done reading cur buf + all copies visible
  }

#pragma unroll
  for (int i = 0; i < 2; i++)
#pragma unroll
    for (int j = 0; j < 4; j++)
#pragma unroll
      for (int r = 0; r < 8; r++) {
        int row = m0 + 32 * waveM + 16 * i + r + 8 * g;
        int col = n0 + 64 * waveN + 16 * j + mr;
        store_out(&C[(size_t)row * N + col], acc[i][j][r]);
      }
}

// --------------------------- per-chunk attention ----------------------------
// One block (128 threads = 4 waves) per (b, chunk, head).
// qkv: [B][T][3072] bf16, q at f=h*64, k at 1024+h*64, v at 2048+h*64.
// y:   [B][T][1024] bf16.  V staged via async global->LDS copies.
__global__ void attn_chunk(const __bf16* __restrict__ qkv,
                           __bf16* __restrict__ y, int NC) {
  const int idx = blockIdx.x;
  const int h = idx % NHEAD;
  const int n = (idx / NHEAD) % NC;
  const int b = idx / (NHEAD * NC);

  __shared__ __attribute__((aligned(32))) __bf16 qs[64][80];
  __shared__ __attribute__((aligned(32))) __bf16 ks[64][80];
  __shared__ __attribute__((aligned(32))) __bf16 vs[64][80];
  __shared__ __attribute__((aligned(32))) float  sc[64][65];

  const int tid = threadIdx.x;
  const size_t baseT = (size_t)b * ((size_t)NC * 64) + (size_t)n * 64;
  const __bf16* qg = qkv + baseT * 3072 + h * 64;
  const __bf16* kg = qg + 1024;
  const __bf16* vg = qg + 2048;

  // stage V: 64 rows x 64 bf16 = 512 b128 chunks, 4 per thread, async
#pragma unroll
  for (int i = 0; i < 4; i++) {
    int c = tid + 128 * i;        // chunk id
    int s = c >> 3, ch = (c & 7) * 8;
    async_copy_b128(vg + (size_t)s * 3072 + ch, &vs[s][ch]);
  }
  // RoPE on Q,K (chunk-local position s, head_dim 64, half=32) -> LDS bf16
  for (int p = tid; p < 64 * 32; p += 128) {
    int s = p >> 5, i = p & 31;
    float ang = (float)s * __powf(10000.0f, -(float)i / 32.0f);
    float c = __cosf(ang), sn = __sinf(ang);
    float q1 = bf2f(qg[(size_t)s * 3072 + i]);
    float q2 = bf2f(qg[(size_t)s * 3072 + i + 32]);
    qs[s][i]      = f2bf(q1 * c + q2 * sn);
    qs[s][i + 32] = f2bf(-q1 * sn + q2 * c);
    float k1 = bf2f(kg[(size_t)s * 3072 + i]);
    float k2 = bf2f(kg[(size_t)s * 3072 + i + 32]);
    ks[s][i]      = f2bf(k1 * c + k2 * sn);
    ks[s][i + 32] = f2bf(-k1 * sn + k2 * c);
  }
  wait_async0();
  __syncthreads();

  const int lane = tid & 31;
  const int w    = tid >> 5;   // wave 0..3 -> q rows 16w..16w+15
  const int m0   = w * 16;
  const int mr   = lane & 15;
  const int g    = lane >> 4;

  // ---- scores = Q K^T ----
  v8f acc[4];
#pragma unroll
  for (int j = 0; j < 4; j++) acc[j] = v8f_zero();
#pragma unroll
  for (int kb = 0; kb < 64; kb += 32) {
    v16bf a;
    {
      v8bf lo = *(const v8bf*)&qs[m0 + mr][kb + 16 * g];
      v8bf hi = *(const v8bf*)&qs[m0 + mr][kb + 8 + 16 * g];
#pragma unroll
      for (int i = 0; i < 8; i++) { a[i] = lo[i]; a[8 + i] = hi[i]; }
    }
#pragma unroll
    for (int j = 0; j < 4; j++) {
      v16bf bb = *(const v16bf*)&ks[16 * j + mr][kb + 16 * g];
      acc[j] = wmma_bf16(a, bb, acc[j]);
    }
  }
  const float scale = 0.125f;   // 1/sqrt(64)
#pragma unroll
  for (int j = 0; j < 4; j++)
#pragma unroll
    for (int r = 0; r < 8; r++)
      sc[m0 + r + 8 * g][16 * j + mr] = acc[j][r] * scale;
  __syncthreads();

  // ---- rowwise softmax over 64 keys (fp32) ----
  if (tid < 64) {
    float mx = -3.4e38f;
    for (int c = 0; c < 64; c++) mx = fmaxf(mx, sc[tid][c]);
    float sum = 0.f;
    for (int c = 0; c < 64; c++) { float e = __expf(sc[tid][c] - mx); sc[tid][c] = e; sum += e; }
    float inv = 1.0f / sum;
    for (int c = 0; c < 64; c++) sc[tid][c] *= inv;
  }
  __syncthreads();

  // ---- y = P V ----
  v8f acc2[4];
#pragma unroll
  for (int j = 0; j < 4; j++) acc2[j] = v8f_zero();
#pragma unroll
  for (int kb = 0; kb < 64; kb += 32) {
    v16bf a;
#pragma unroll
    for (int i = 0; i < 8; i++) {
      a[i]     = f2bf(sc[m0 + mr][kb + 16 * g + i]);
      a[8 + i] = f2bf(sc[m0 + mr][kb + 8 + 16 * g + i]);
    }
#pragma unroll
    for (int j = 0; j < 4; j++) {
      v16bf bb;
#pragma unroll
      for (int jj = 0; jj < 16; jj++) bb[jj] = vs[kb + 16 * g + jj][16 * j + mr];
      acc2[j] = wmma_bf16(a, bb, acc2[j]);
    }
  }
  __bf16* yg = y + baseT * 1024 + h * 64;
#pragma unroll
  for (int j = 0; j < 4; j++)
#pragma unroll
    for (int r = 0; r < 8; r++) {
      int s = m0 + r + 8 * g;
      int dd = 16 * j + mr;
      yg[(size_t)s * 1024 + dd] = f2bf(acc2[j][r]);
    }
}

// ------------------------------- launch -------------------------------------
extern "C" void kernel_launch(void* const* d_in, const int* in_sizes, int n_in,
                              void* d_out, int out_size, void* d_ws, size_t ws_size,
                              hipStream_t stream) {
  const float* x      = (const float*)d_in[0];   // [4,4096,1024]
  const float* w_attn = (const float*)d_in[1];   // [3072,1024]
  const float* w_proj = (const float*)d_in[2];   // [1024,1024]
  float* out = (float*)d_out;                    // [4,4096,1024] fp32

  const int Bb = 4, T = 4096, C = NEMBD, NC = T / CHUNK;
  const size_t MT = (size_t)Bb * T;              // 16384 rows

  // workspace layout (bf16 staging), all 32B aligned
  char* ws = (char*)d_ws;
  size_t o = 0;
  __bf16* x_bf  = (__bf16*)(ws + o); o += MT * C * 2;               // 32 MB
  __bf16* wa_bf = (__bf16*)(ws + o); o += (size_t)3 * C * C * 2;    // 6 MB
  __bf16* wp_bf = (__bf16*)(ws + o); o += (size_t)C * C * 2;        // 2 MB
  __bf16* qkvbf = (__bf16*)(ws + o); o += MT * 3 * C * 2;           // 96 MB
  __bf16* y_bf  = (__bf16*)(ws + o);                                // 32 MB

  // fp32 -> bf16
  cvt_f32_bf16<<<2048, 256, 0, stream>>>(x, x_bf, (int)(MT * C));
  cvt_f32_bf16<<<512,  256, 0, stream>>>(w_attn, wa_bf, 3 * C * C);
  cvt_f32_bf16<<<256,  256, 0, stream>>>(w_proj, wp_bf, C * C);

  // qkv = x @ w_attn^T : M=16384 N=3072 K=1024
  gemm_bf16_wmma<__bf16><<<(16384 / 128) * (3072 / 128), 256, 0, stream>>>(
      x_bf, wa_bf, qkvbf, 16384, 3072, 1024);

  // per-chunk attention (rope + softmax(QK^T/8)V)
  attn_chunk<<<Bb * NC * NHEAD, 128, 0, stream>>>(qkvbf, y_bf, NC);

  // out = y @ w_proj^T : M=16384 N=1024 K=1024, fp32 out
  gemm_bf16_wmma<float><<<(16384 / 128) * (1024 / 128), 256, 0, stream>>>(
      y_bf, wp_bf, out, 16384, 1024, 1024);
}
